// Decoder_TRANSFORMERREEMB_42743514530435
// MI455X (gfx1250) — matrix operand order repaired and checked by reading, not compiled, not run on hardware
//
#include <hip/hip_runtime.h>
#include <math.h>

// ---------------------------------------------------------------------------
// Problem constants (from the reference)
// ---------------------------------------------------------------------------
#define B_SZ    8
#define N_SEQ   2048
#define POSE_DM 64
#define A_LATD  256
#define POS_DIM 7
#define HEADS_N 4
#define DH_TMP  32      // temporal attention head dim (HID=128)
#define DH_DEC  16      // decoder attention head dim  (E=64)
#define LAYERS_N 4
#define NBUCK   32
#define R_TOT   (B_SZ * N_SEQ)   // 16384 rows

// Set to 0 to fall back to regular LDS staging if the async mnemonic is
// rejected by the assembler.
#define USE_ASYNC_COPY 1

typedef __attribute__((ext_vector_type(16))) _Float16 v16h;
typedef __attribute__((ext_vector_type(8)))  _Float16 v8h;
typedef __attribute__((ext_vector_type(8)))  float    v8f;

__device__ __forceinline__ v16h v16_combine(v8h lo, v8h hi) {
  return __builtin_shufflevector(lo, hi, 0, 1, 2, 3, 4, 5, 6, 7,
                                 8, 9, 10, 11, 12, 13, 14, 15);
}
__device__ __forceinline__ v8h v8h_zero() {
  v8h z;
#pragma unroll
  for (int i = 0; i < 8; ++i) z[i] = (_Float16)0.f;
  return z;
}
__device__ __forceinline__ v8f v8f_zero() {
  v8f z;
#pragma unroll
  for (int i = 0; i < 8; ++i) z[i] = 0.f;
  return z;
}

// 16-byte global -> LDS copy. Async variant uses the CDNA5 async DMA path
// (ASYNCcnt-tracked); flat LDS pointers carry the LDS byte address in their
// low 32 bits (ISA aperture rules), which is what VDST wants.
__device__ __forceinline__ void lds_copy16(void* lptr, const void* gptr) {
#if USE_ASYNC_COPY
  unsigned l = (unsigned)(unsigned long long)lptr;
  unsigned long long g = (unsigned long long)gptr;
  asm volatile("global_load_async_to_lds_b128 %0, %1, off"
               :: "v"(l), "v"(g) : "memory");
#else
  *(v8h*)lptr = *(const v8h*)gptr;
#endif
}
__device__ __forceinline__ void lds_copy_fence() {
#if USE_ASYNC_COPY
  asm volatile("s_wait_asynccnt 0" ::: "memory");
#endif
}

// ---------------------------------------------------------------------------
// f16 WMMA GEMM:  C[M,Nc] = A[M,K] * W[Nc,K]^T (+bias[col])
// A row-major f16 (lda); W row-major f16 (ldw) == the original weight layout,
// which is exactly the column-major B operand -> all LDS accesses contiguous.
// Double-buffered async LDS staging: tile i+1 DMA overlaps tile i WMMAs.
// M % 128 == 0, Nc % 64 == 0, K % 32 == 0 (true for every GEMM issued).
// Block = 256 threads = 8 waves; wave (wm,wn) owns a 32x32 sub-tile.
// ---------------------------------------------------------------------------
template <bool OUT16, bool HASBIAS>
__global__ void __launch_bounds__(256) k_gemm_f16(
    const _Float16* __restrict__ A, int lda,
    const _Float16* __restrict__ W, int ldw,
    float* __restrict__ C, int ldc,
    _Float16* __restrict__ C16, int ldc16,
    const float* __restrict__ bias,
    int M, int Nc, int K)
{
  __shared__ alignas(16) _Float16 As[2][128][40];  // 80B stride (16B multiple)
  __shared__ alignas(16) _Float16 Ws[2][64][40];

  const int tid = threadIdx.x, lane = tid & 31, wv = tid >> 5;
  const int wm = wv >> 1, wn = wv & 1;
  const int m0 = blockIdx.x * 128, n0 = blockIdx.y * 64;

  v8f acc[2][2];
#pragma unroll
  for (int i = 0; i < 2; ++i)
#pragma unroll
    for (int j = 0; j < 2; ++j) acc[i][j] = v8f_zero();

  const int arow = tid >> 1, acb = (tid & 1) * 16;  // 2 thr/row, 2x 16B each
  const int wrow = tid >> 2, wcb = (tid & 3) * 8;   // 4 thr/row, 1x 16B each
  const _Float16* Ag = &A[(size_t)(m0 + arow) * lda + acb];
  const _Float16* Wg = &W[(size_t)(n0 + wrow) * ldw + wcb];

  auto stage = [&](int bufi, int k0) {
    lds_copy16(&As[bufi][arow][acb],     Ag + k0);
    lds_copy16(&As[bufi][arow][acb + 8], Ag + k0 + 8);
    lds_copy16(&Ws[bufi][wrow][wcb],     Wg + k0);
  };

  stage(0, 0);
  lds_copy_fence();
  __syncthreads();

  int bufi = 0;
  for (int k0 = 0; k0 < K; k0 += 32) {
    if (k0 + 32 < K) stage(bufi ^ 1, k0 + 32);  // overlap DMA with WMMAs

    const int kh = (lane >> 4) * 8;    // A-layout K half select
    const int kb = (lane >> 4) * 16;   // contiguous-order K half select (unused)
    v16h af[2], bf[2];
#pragma unroll
    for (int ri = 0; ri < 2; ++ri) {
      const int r = wm * 32 + ri * 16 + (lane & 15);
      af[ri] = v16_combine(*(const v8h*)&As[bufi][r][kh],
                           *(const v8h*)&As[bufi][r][kh + 16]);
    }
#pragma unroll
    for (int ci = 0; ci < 2; ++ci) {
      const int cc = wn * 32 + ci * 16 + (lane & 15);
      bf[ci] = v16_combine(*(const v8h*)&Ws[bufi][cc][kh],
                           *(const v8h*)&Ws[bufi][cc][kh + 16]);
    }
#pragma unroll
    for (int ri = 0; ri < 2; ++ri)
#pragma unroll
      for (int ci = 0; ci < 2; ++ci)
        acc[ri][ci] = __builtin_amdgcn_wmma_f32_16x16x32_f16(
            false, af[ri], false, bf[ci], (short)0, acc[ri][ci], false, false);

    lds_copy_fence();
    __syncthreads();
    bufi ^= 1;
  }

  const int rhi = (lane >> 4) ? 8 : 0;
#pragma unroll
  for (int ri = 0; ri < 2; ++ri)
#pragma unroll
    for (int ci = 0; ci < 2; ++ci) {
      const int col = n0 + wn * 32 + ci * 16 + (lane & 15);
      float bv = 0.f;
      if constexpr (HASBIAS) bv = bias[col];
#pragma unroll
      for (int r = 0; r < 8; ++r) {
        const int row = m0 + wm * 32 + ri * 16 + r + rhi;
        const float v = acc[ri][ci][r] + bv;
        C[(size_t)row * ldc + col] = v;
        if constexpr (OUT16) C16[(size_t)row * ldc16 + col] = (_Float16)v;
      }
    }
}

// ---------------------------------------------------------------------------
// Flash attention, fully transposed so every LDS access is a contiguous b128:
//   S^T = K . Q^T   (keys = WMMA rows, queries = WMMA cols)
//   O^T = V^T . P^T (head-dims = rows, queries = cols)
// Q,K f16 (B,H,N,DH); V f16 transposed (B,H,DH,N); O f16 (B,N,H*DH).
// Double-buffered async K/V staging; block = 128 threads = 4 waves; wave owns
// 16 query columns; 64-key tiles. Softmax stats live per lane column
// (partner = lane^16 -> single shfl).
// ---------------------------------------------------------------------------
template <int DH, bool USE_BIAS, bool USE_MASK>
__global__ void __launch_bounds__(128) k_attn(
    const _Float16* __restrict__ Qm, const _Float16* __restrict__ Km,
    const _Float16* __restrict__ Vm, _Float16* __restrict__ Om,
    const float* __restrict__ btab, const unsigned char* __restrict__ maskv,
    int Bn, int N, int H)
{
  constexpr int KT = 64;
  constexpr int NT = DH / 16;
  constexpr int KROW = DH + 8;          // 40 or 24 halves: 16B-multiple stride
  __shared__ alignas(16) _Float16 Ks [2][KT][KROW];
  __shared__ alignas(16) _Float16 VsT[2][DH][KT + 8];
  __shared__ alignas(16) _Float16 Pt [4][16][KT + 8];

  const int b = blockIdx.z, h = blockIdx.y;
  const int q0 = blockIdx.x * 64;
  const int tid = threadIdx.x, lane = tid & 31, wv = tid >> 5;
  const int rhi = (lane >> 4) ? 8 : 0;
  const int qcol = lane & 15;
  const int qglob = q0 + wv * 16 + qcol;
  const int kh = (lane >> 4) * 8;
  const int kb = (lane >> 4) * 16;

  const _Float16* Qb = Qm + (size_t)(b * H + h) * N * DH;
  const _Float16* Kb = Km + (size_t)(b * H + h) * N * DH;
  const _Float16* Vb = Vm + (size_t)(b * H + h) * DH * N;

  // Q^T B-fragment (vectorized): lane col = query, elements = head dim
  v16h bq;
  if (DH == 32) {
    bq = v16_combine(*(const v8h*)&Qb[(size_t)qglob * DH + kh],
                     *(const v8h*)&Qb[(size_t)qglob * DH + kh + 16]);
  } else {
    bq = v16_combine(*(const v8h*)&Qb[(size_t)qglob * DH + kh], v8h_zero());
  }

  v8f accO[NT];
#pragma unroll
  for (int t = 0; t < NT; ++t) accO[t] = v8f_zero();
  float m_run = -INFINITY, l_run = 0.f;

  auto stage = [&](int bufi, int kt) {
    for (int i = tid; i < KT * (DH / 8); i += 128) {
      const int kr = i / (DH / 8), c8 = (i % (DH / 8)) * 8;
      lds_copy16(&Ks[bufi][kr][c8], &Kb[(size_t)(kt * KT + kr) * DH + c8]);
    }
    for (int i = tid; i < DH * (KT / 8); i += 128) {
      const int d = i / (KT / 8), c8 = (i % (KT / 8)) * 8;
      lds_copy16(&VsT[bufi][d][c8], &Vb[(size_t)d * N + kt * KT + c8]);
    }
  };

  stage(0, 0);
  lds_copy_fence();
  __syncthreads();

  int bufi = 0;
  const int nkt = N / KT;
  for (int kt = 0; kt < nkt; ++kt) {
    if (kt + 1 < nkt) stage(bufi ^ 1, kt + 1);  // overlap DMA with compute

    // S^T: 4 key sub-tiles of 16
    v8f ST[4];
#pragma unroll
    for (int c = 0; c < 4; ++c) {
      const int krow = c * 16 + (lane & 15);
      v8h lo = *(const v8h*)&Ks[bufi][krow][kh];
      v8h hi = (DH == 32) ? *(const v8h*)&Ks[bufi][krow][kh + 16] : v8h_zero();
      ST[c] = __builtin_amdgcn_wmma_f32_16x16x32_f16(
          false, v16_combine(lo, hi), false, bq, (short)0, v8f_zero(),
          false, false);
    }

    if (USE_BIAS || USE_MASK) {
#pragma unroll
      for (int c = 0; c < 4; ++c)
#pragma unroll
        for (int r = 0; r < 8; ++r) {
          const int j_ = kt * KT + c * 16 + r + rhi;   // key index
          float s = ST[c][r];
          if (USE_BIAS)
            s += btab[(size_t)h * (2 * N - 1) + (j_ - qglob) + (N - 1)];
          if (USE_MASK && !maskv[(size_t)b * N + j_]) s = -1e9f;
          ST[c][r] = s;
        }
    }

    // online softmax per query column (lane + lane^16 hold the 64 keys)
    float mp = -INFINITY;
#pragma unroll
    for (int c = 0; c < 4; ++c)
#pragma unroll
      for (int r = 0; r < 8; ++r) mp = fmaxf(mp, ST[c][r]);
    mp = fmaxf(mp, __shfl_xor(mp, 16, 32));
    const float mn = fmaxf(m_run, mp);
    const float sc = __expf(m_run - mn);
    float rs = 0.f;
#pragma unroll
    for (int c = 0; c < 4; ++c) {
      v8h ph;
#pragma unroll
      for (int r = 0; r < 8; ++r) {
        const float p = __expf(ST[c][r] - mn);
        rs += p;
        ph[r] = (_Float16)p;
      }
      // 8 consecutive keys of this query column -> one b128 store
      *(v8h*)&Pt[wv][qcol][c * 16 + rhi] = ph;
    }
    rs += __shfl_xor(rs, 16, 32);
    l_run = l_run * sc + rs;
    m_run = mn;
#pragma unroll
    for (int t = 0; t < NT; ++t)
#pragma unroll
      for (int e = 0; e < 8; ++e) accO[t][e] *= sc;

    // O^T += V^T . P^T  (two K=32 chunks over the 64 keys)
#pragma unroll
    for (int ch = 0; ch < 2; ++ch) {
#pragma unroll
      for (int t = 0; t < NT; ++t) {
        const int drow = t * 16 + (lane & 15);
        v16h aV = v16_combine(*(const v8h*)&VsT[bufi][drow][ch * 32 + kh],
                              *(const v8h*)&VsT[bufi][drow][ch * 32 + kh + 16]);
        v16h bP = v16_combine(*(const v8h*)&Pt[wv][qcol][ch * 32 + kh],
                              *(const v8h*)&Pt[wv][qcol][ch * 32 + kh + 16]);
        accO[t] = __builtin_amdgcn_wmma_f32_16x16x32_f16(
            false, aV, false, bP, (short)0, accO[t], false, false);
      }
    }

    lds_copy_fence();
    __syncthreads();
    bufi ^= 1;
  }

  // epilogue: contiguous along head dim for each lane -> one b128 store per t
  const float inv = (l_run > 0.f) ? 1.f / l_run : 0.f;
#pragma unroll
  for (int t = 0; t < NT; ++t) {
    v8h oh;
#pragma unroll
    for (int r = 0; r < 8; ++r) oh[r] = (_Float16)(accO[t][r] * inv);
    *(v8h*)&Om[(size_t)(b * N + qglob) * (H * DH) + h * DH + t * 16 + rhi] = oh;
  }
}

// ---------------------------------------------------------------------------
// Elementwise / small helper kernels
// ---------------------------------------------------------------------------
__global__ void k_cast_f16(const float* __restrict__ s, _Float16* __restrict__ d, int n)
{
  const int i = blockIdx.x * 256 + threadIdx.x;
  if (i < n) d[i] = (_Float16)s[i];
}

// relative-position bias table: tab[h, rel + N-1], rel = j - i
__global__ void k_bias_tab(const float* __restrict__ emb, float* __restrict__ tab)
{
  const int W = 2 * N_SEQ - 1;
  const int idx = blockIdx.x * 256 + threadIdx.x;
  if (idx >= HEADS_N * W) return;
  const int h = idx / W;
  const int rel = idx % W - (N_SEQ - 1);
  const int neg = -rel;
  const int half = NBUCK / 2;        // 16
  const int max_exact = half / 2;    // 8
  const int ret = (neg < 0) ? half : 0;
  const int na = (neg < 0) ? -neg : neg;
  int bucket;
  if (na < max_exact) {
    bucket = ret + na;
  } else {
    const float naf = (float)(na < 1 ? 1 : na);
    int vl = max_exact + (int)(logf(naf / (float)max_exact) / logf(4.f) *
                               (float)(half - max_exact));
    vl = vl < (half - 1) ? vl : (half - 1);
    bucket = ret + vl;
  }
  const float win = (rel > 32 || rel < -32) ? -1e8f : 0.f;
  tab[idx] = emb[bucket * HEADS_N + h] + win;
}

// sinusoidal PE table -> f16 (A operand of the ip projection GEMM)
__global__ void k_pe(_Float16* __restrict__ pe)
{
  const int idx = blockIdx.x * 256 + threadIdx.x;
  if (idx >= N_SEQ * POSE_DM) return;
  const int n = idx / POSE_DM, d = idx % POSE_DM;
  const int dd = d & ~1;
  const float div = __expf((float)dd * (-logf(10000.f) / (float)POSE_DM));
  const float a = (float)n * div;
  pe[idx] = (_Float16)((d & 1) ? cosf(a) : sinf(a));
}

// x_ref row per batch: xr[b][d] = fp_w[d,:] . x[b,0,:] + fp_b[d]
__global__ void k_xref(const float* __restrict__ x, const float* __restrict__ w,
                       const float* __restrict__ b, float* __restrict__ xr)
{
  const int bb = blockIdx.x, d = threadIdx.x;
  float s = b[d];
  for (int j = 0; j < POS_DIM; ++j)
    s += w[d * POS_DIM + j] * x[(size_t)bb * N_SEQ * POS_DIM + j];
  xr[bb * POSE_DM + d] = s;
}

// zc_in f16 (R x 576): [x_ref(b) | z(n,b) | ye(b,n)]
__global__ void k_build_zc(const float* __restrict__ xr, const float* __restrict__ z,
                           const float* __restrict__ ye, _Float16* __restrict__ zc)
{
  const size_t idx = (size_t)blockIdx.x * 256 + threadIdx.x;
  const size_t tot = (size_t)R_TOT * 576;
  if (idx >= tot) return;
  const int col = (int)(idx % 576);
  const size_t r = idx / 576;
  const int b = (int)(r / N_SEQ), n = (int)(r % N_SEQ);
  float v;
  if (col < 64)       v = xr[b * POSE_DM + col];
  else if (col < 320) v = z[((size_t)n * B_SZ + b) * A_LATD + (col - 64)];
  else                v = ye[r * A_LATD + (col - 320)];
  zc[idx] = (_Float16)v;
}

// layernorm (optional residual / optional beta), writes f32 and/or f16
__global__ void k_ln(const float* __restrict__ x, const float* __restrict__ res,
                     const float* __restrict__ w, const float* __restrict__ beta,
                     float* __restrict__ y32, _Float16* __restrict__ y16, int rows)
{
  const int r = blockIdx.x * 256 + threadIdx.x;
  if (r >= rows) return;
  const float* xr = x + (size_t)r * POSE_DM;
  const float* rr = res ? res + (size_t)r * POSE_DM : nullptr;
  float mean = 0.f;
  for (int d = 0; d < POSE_DM; ++d) mean += xr[d] + (rr ? rr[d] : 0.f);
  mean *= (1.f / POSE_DM);
  float var = 0.f;
  for (int d = 0; d < POSE_DM; ++d) {
    const float t = xr[d] + (rr ? rr[d] : 0.f) - mean;
    var += t * t;
  }
  const float inv = rsqrtf(var * (1.f / POSE_DM) + 1e-5f);
  for (int d = 0; d < POSE_DM; ++d) {
    const float v = (xr[d] + (rr ? rr[d] : 0.f) - mean) * inv * w[d] +
                    (beta ? beta[d] : 0.f);
    if (y32) y32[(size_t)r * POSE_DM + d] = v;
    if (y16) y16[(size_t)r * POSE_DM + d] = (_Float16)v;
  }
}

// split rows of a QKV-style f32 buffer into f16 heads:
//   vtrans==0 -> (B,H,N,DH)   (Q/K layout)
//   vtrans==1 -> (B,H,DH,N)   (V layout, feeds contiguous V^T fragments)
// Optional query scale and ROT_DIM=4 rotary embedding.
__global__ void k_to_heads(const float* __restrict__ src, int ld, int coloff,
                           _Float16* __restrict__ dst, float scale, int do_rot,
                           int vtrans, int Bn, int N, int H, int DH)
{
  const size_t idx = (size_t)blockIdx.x * 256 + threadIdx.x;
  const size_t tot = (size_t)Bn * N * H * DH;
  if (idx >= tot) return;
  const int d = (int)(idx % DH);
  const int n = (int)((idx / DH) % N);
  const int h = (int)((idx / ((size_t)DH * N)) % H);
  const int b = (int)(idx / ((size_t)DH * N * H));
  const size_t rbase = (size_t)(b * N + n) * ld + coloff + h * DH;
  float v = src[rbase + d] * scale;
  if (do_rot && d < 4) {
    const float vp = src[rbase + (d ^ 1)] * scale;
    const float f = (d < 2) ? (float)n : 0.01f * (float)n;
    const float c = cosf(f), s = sinf(f);
    v = (d & 1) ? (v * c + vp * s) : (v * c - vp * s);
  }
  const size_t di = vtrans
      ? (((size_t)(b * H + h) * DH + d) * N + n)
      : (((size_t)(b * H + h) * N + n) * DH + d);
  dst[di] = (_Float16)v;
}

// out[b,n] = tqp[n] + proj[n]  (temporal attention is batch-invariant)
__global__ void k_bcast_add(const float* __restrict__ tqp, const float* __restrict__ proj,
                            float* __restrict__ o32, _Float16* __restrict__ o16)
{
  const size_t idx = (size_t)blockIdx.x * 256 + threadIdx.x;
  if (idx >= (size_t)R_TOT * POSE_DM) return;
  const int d = (int)(idx % POSE_DM);
  const int n = (int)((idx / POSE_DM) % N_SEQ);
  const float v = tqp[(size_t)n * POSE_DM + d] + proj[(size_t)n * POSE_DM + d];
  o32[idx] = v;
  o16[idx] = (_Float16)v;
}

__global__ void k_gelu(const float* __restrict__ x, _Float16* __restrict__ g, size_t n)
{
  const size_t i = (size_t)blockIdx.x * 256 + threadIdx.x;
  if (i >= n) return;
  const float v = x[i];
  g[i] = (_Float16)(0.5f * v * (1.f + erff(v * 0.70710678f)));
}

// final 64 -> 7 projection + mask, directly to output (B,N,7) f32
__global__ void k_final(const float* __restrict__ o, const float* __restrict__ w,
                        const float* __restrict__ b, const unsigned char* __restrict__ m,
                        float* __restrict__ out)
{
  const size_t idx = (size_t)blockIdx.x * 256 + threadIdx.x;
  if (idx >= (size_t)R_TOT * POS_DIM) return;
  const int p = (int)(idx % POS_DIM);
  const size_t r = idx / POS_DIM;
  float s = 0.f;
  if (m[r]) {
    s = b[p];
    for (int d = 0; d < POSE_DM; ++d)
      s += o[r * POSE_DM + d] * w[p * POSE_DM + d];
  }
  out[idx] = s;
}

// ---------------------------------------------------------------------------
// Host orchestration
// ---------------------------------------------------------------------------
extern "C" void kernel_launch(void* const* d_in, const int* in_sizes, int n_in,
                              void* d_out, int out_size, void* d_ws, size_t ws_size,
                              hipStream_t stream)
{
  (void)in_sizes; (void)n_in; (void)out_size; (void)ws_size;
  const float* x     = (const float*)d_in[0];
  const float* z     = (const float*)d_in[1];
  const float* y     = (const float*)d_in[2];
  const unsigned char* mask = (const unsigned char*)d_in[3];
  const float* fp_w  = (const float*)d_in[5];
  const float* fp_b  = (const float*)d_in[6];
  const float* ae_w  = (const float*)d_in[7];
  const float* ae_b  = (const float*)d_in[8];
  const float* zt_w  = (const float*)d_in[9];
  const float* zt_b  = (const float*)d_in[10];
  const float* ip_w  = (const float*)d_in[11];
  const float* ip_b  = (const float*)d_in[12];
  const float* rp_emb   = (const float*)d_in[13];
  const float* pn_gamma = (const float*)d_in[14];
  const float* qkv_w = (const float*)d_in[15];
  const float* ao_w  = (const float*)d_in[16];
  const float* sa_in_w  = (const float*)d_in[17];
  const float* sa_in_b  = (const float*)d_in[18];
  const float* sa_out_w = (const float*)d_in[19];
  const float* sa_out_b = (const float*)d_in[20];
  const float* ca_in_w  = (const float*)d_in[21];
  const float* ca_in_b  = (const float*)d_in[22];
  const float* ca_out_w = (const float*)d_in[23];
  const float* ca_out_b = (const float*)d_in[24];
  const float* l1_w  = (const float*)d_in[25];
  const float* l1_b  = (const float*)d_in[26];
  const float* l2_w  = (const float*)d_in[27];
  const float* l2_b  = (const float*)d_in[28];
  const float* ln_w  = (const float*)d_in[29];
  const float* ln_b  = (const float*)d_in[30];
  const float* fin_w = (const float*)d_in[31];
  const float* fin_b = (const float*)d_in[32];
  float* outp = (float*)d_out;

  // ---- bump allocator over d_ws ----
  char* base = (char*)d_ws;
  size_t off = 0;
  auto alloc = [&](size_t bytes) -> void* {
    off = (off + 255) & ~(size_t)255;
    void* p = base + off;
    off += bytes;
    return p;
  };
  auto aF16 = [&](size_t n) { return (_Float16*)alloc(n * sizeof(_Float16)); };
  auto aF32 = [&](size_t n) { return (float*)alloc(n * sizeof(float)); };

  // f16 copies of weights, KEPT IN ORIGINAL (out,in) ROW-MAJOR: that layout is
  // directly the column-major B operand of the WMMA GEMM.
  _Float16* ae16  = aF16((size_t)256 * 1024);
  _Float16* zt16  = aF16((size_t)64 * 576);
  _Float16* ip16  = aF16((size_t)64 * 64);
  _Float16* qkv16 = aF16((size_t)384 * 64);
  _Float16* ao16  = aF16((size_t)64 * 128);
  _Float16 *sa16[LAYERS_N], *sao16[LAYERS_N], *ca16[LAYERS_N], *cao16[LAYERS_N],
           *l116[LAYERS_N], *l216[LAYERS_N];
  for (int l = 0; l < LAYERS_N; ++l) {
    sa16[l]  = aF16((size_t)192 * 64);
    sao16[l] = aF16((size_t)64 * 64);
    ca16[l]  = aF16((size_t)192 * 64);
    cao16[l] = aF16((size_t)64 * 64);
    l116[l]  = aF16((size_t)1024 * 64);
    l216[l]  = aF16((size_t)64 * 1024);
  }

  float*    btab   = aF32((size_t)HEADS_N * (2 * N_SEQ - 1));
  float*    xref   = aF32((size_t)B_SZ * POSE_DM);
  _Float16* pe16   = aF16((size_t)N_SEQ * POSE_DM);
  float*    tqp32  = aF32((size_t)N_SEQ * POSE_DM);
  _Float16* xn16   = aF16((size_t)N_SEQ * POSE_DM);
  _Float16* tq_q16 = aF16((size_t)N_SEQ * 128);
  _Float16* tq_k16 = aF16((size_t)N_SEQ * 128);
  _Float16* tq_v16 = aF16((size_t)N_SEQ * 128);
  _Float16* tO16   = aF16((size_t)N_SEQ * 128);
  float*    tproj  = aF32((size_t)N_SEQ * POSE_DM);
  float*    mem32  = aF32((size_t)R_TOT * POSE_DM);
  _Float16* mem16  = aF16((size_t)R_TOT * POSE_DM);
  float*    out32  = aF32((size_t)R_TOT * POSE_DM);
  _Float16* out16  = aF16((size_t)R_TOT * POSE_DM);
  float*    delta  = aF32((size_t)R_TOT * POSE_DM);
  _Float16* q16    = aF16((size_t)R_TOT * 64);
  _Float16* k16    = aF16((size_t)R_TOT * 64);
  _Float16* v16    = aF16((size_t)R_TOT * 64);
  _Float16* aO16   = aF16((size_t)R_TOT * 64);
  float*    SCR32  = aF32((size_t)R_TOT * 1024);   // ye / qkv / h1
  _Float16* SCR16  = aF16((size_t)R_TOT * 1024);   // y16 / zc16 / gelu16

  auto g1 = [](size_t n) { return dim3((unsigned)((n + 255) / 256)); };
  auto cast = [&](const float* s, _Float16* d, int n) {
    k_cast_f16<<<g1((size_t)n), 256, 0, stream>>>(s, d, n);
  };
  auto gemm = [&](const _Float16* A, int lda, const _Float16* W, int ldw,
                  float* C, int ldc, _Float16* C16, int ldc16,
                  const float* bias, int M, int Nc, int K) {
    dim3 grid(M / 128, Nc / 64);
    if (C16 && bias)
      k_gemm_f16<true, true><<<grid, 256, 0, stream>>>(
          A, lda, W, ldw, C, ldc, C16, ldc16, bias, M, Nc, K);
    else if (bias)
      k_gemm_f16<false, true><<<grid, 256, 0, stream>>>(
          A, lda, W, ldw, C, ldc, C16, ldc16, bias, M, Nc, K);
    else
      k_gemm_f16<false, false><<<grid, 256, 0, stream>>>(
          A, lda, W, ldw, C, ldc, C16, ldc16, bias, M, Nc, K);
  };
  auto heads = [&](const float* src, int ld, int coloff, _Float16* dst,
                   float sc, int rot, int vt, int Bn, int H, int DH) {
    k_to_heads<<<g1((size_t)Bn * N_SEQ * H * DH), 256, 0, stream>>>(
        src, ld, coloff, dst, sc, rot, vt, Bn, N_SEQ, H, DH);
  };

  // ---- weight casts + bias table ----
  cast(ae_w, ae16, 256 * 1024);
  cast(zt_w, zt16, 64 * 576);
  cast(ip_w, ip16, 64 * 64);
  cast(qkv_w, qkv16, 384 * 64);
  cast(ao_w, ao16, 64 * 128);
  for (int l = 0; l < LAYERS_N; ++l) {
    cast(sa_in_w  + (size_t)l * 192 * 64,  sa16[l],  192 * 64);
    cast(sa_out_w + (size_t)l * 64 * 64,   sao16[l], 64 * 64);
    cast(ca_in_w  + (size_t)l * 192 * 64,  ca16[l],  192 * 64);
    cast(ca_out_w + (size_t)l * 64 * 64,   cao16[l], 64 * 64);
    cast(l1_w     + (size_t)l * 1024 * 64, l116[l],  1024 * 64);
    cast(l2_w     + (size_t)l * 64 * 1024, l216[l],  64 * 1024);
  }
  k_bias_tab<<<g1((size_t)HEADS_N * (2 * N_SEQ - 1)), 256, 0, stream>>>(rp_emb, btab);

  // ---- memory path: ye = y @ ae_w.T + b ; zc ; memory = zc @ zt_w.T + b ----
  cast(y, SCR16, R_TOT * 1024);
  gemm(SCR16, 1024, ae16, 1024, SCR32, 256, nullptr, 0, ae_b, R_TOT, 256, 1024);
  k_xref<<<dim3(B_SZ), dim3(POSE_DM), 0, stream>>>(x, fp_w, fp_b, xref);
  k_build_zc<<<g1((size_t)R_TOT * 576), 256, 0, stream>>>(xref, z, SCR32, SCR16);
  gemm(SCR16, 576, zt16, 576, mem32, 64, mem16, 64, zt_b, R_TOT, 64, 576);

  // ---- temporal attention (batch-invariant: computed once over N rows) ----
  k_pe<<<g1((size_t)N_SEQ * POSE_DM), 256, 0, stream>>>(pe16);
  gemm(pe16, 64, ip16, 64, tqp32, 64, nullptr, 0, ip_b, N_SEQ, 64, 64);
  k_ln<<<g1(N_SEQ), 256, 0, stream>>>(tqp32, nullptr, pn_gamma, nullptr,
                                      nullptr, xn16, N_SEQ);
  gemm(xn16, 64, qkv16, 64, SCR32, 384, nullptr, 0, nullptr, N_SEQ, 384, 64);
  heads(SCR32, 384, 0,   tq_q16, 0.17677669529f, 1, 0, 1, HEADS_N, DH_TMP);
  heads(SCR32, 384, 128, tq_k16, 1.f,            1, 0, 1, HEADS_N, DH_TMP);
  heads(SCR32, 384, 256, tq_v16, 1.f,            0, 1, 1, HEADS_N, DH_TMP);
  {
    dim3 grid(N_SEQ / 64, HEADS_N, 1);
    k_attn<DH_TMP, true, false><<<grid, 128, 0, stream>>>(
        tq_q16, tq_k16, tq_v16, tO16, btab, nullptr, 1, N_SEQ, HEADS_N);
  }
  gemm(tO16, 128, ao16, 128, tproj, 64, nullptr, 0, nullptr, N_SEQ, 64, 128);
  k_bcast_add<<<g1((size_t)R_TOT * POSE_DM), 256, 0, stream>>>(tqp32, tproj,
                                                               out32, out16);

  // ---- decoder layers ----
  const dim3 agrid(N_SEQ / 64, HEADS_N, B_SZ);
  for (int l = 0; l < LAYERS_N; ++l) {
    // self-attention
    gemm(out16, 64, sa16[l], 64, SCR32, 192, nullptr, 0,
         sa_in_b + (size_t)l * 192, R_TOT, 192, 64);
    heads(SCR32, 192, 0,   q16, 0.25f, 0, 0, B_SZ, HEADS_N, DH_DEC);
    heads(SCR32, 192, 64,  k16, 1.f,   0, 0, B_SZ, HEADS_N, DH_DEC);
    heads(SCR32, 192, 128, v16, 1.f,   0, 1, B_SZ, HEADS_N, DH_DEC);
    k_attn<DH_DEC, true, true><<<agrid, 128, 0, stream>>>(
        q16, k16, v16, aO16, btab, mask, B_SZ, N_SEQ, HEADS_N);
    gemm(aO16, 64, sao16[l], 64, delta, 64, nullptr, 0,
         sa_out_b + (size_t)l * 64, R_TOT, 64, 64);
    k_ln<<<g1(R_TOT), 256, 0, stream>>>(out32, delta,
        ln_w + (size_t)(l * 3 + 0) * 64, ln_b + (size_t)(l * 3 + 0) * 64,
        out32, out16, R_TOT);

    // cross-attention (K/V from memory); weight rows 0..63 = Wq, 64..191 = Wk|Wv
    gemm(out16, 64, ca16[l], 64, SCR32, 64, nullptr, 0,
         ca_in_b + (size_t)l * 192, R_TOT, 64, 64);
    heads(SCR32, 64, 0, q16, 0.25f, 0, 0, B_SZ, HEADS_N, DH_DEC);
    gemm(mem16, 64, ca16[l] + (size_t)64 * 64, 64, SCR32, 128, nullptr, 0,
         ca_in_b + (size_t)l * 192 + 64, R_TOT, 128, 64);
    heads(SCR32, 128, 0,  k16, 1.f, 0, 0, B_SZ, HEADS_N, DH_DEC);
    heads(SCR32, 128, 64, v16, 1.f, 0, 1, B_SZ, HEADS_N, DH_DEC);
    k_attn<DH_DEC, false, false><<<agrid, 128, 0, stream>>>(
        q16, k16, v16, aO16, nullptr, nullptr, B_SZ, N_SEQ, HEADS_N);
    gemm(aO16, 64, cao16[l], 64, delta, 64, nullptr, 0,
         ca_out_b + (size_t)l * 64, R_TOT, 64, 64);
    k_ln<<<g1(R_TOT), 256, 0, stream>>>(out32, delta,
        ln_w + (size_t)(l * 3 + 1) * 64, ln_b + (size_t)(l * 3 + 1) * 64,
        out32, out16, R_TOT);

    // feed-forward
    gemm(out16, 64, l116[l], 64, SCR32, 1024, nullptr, 0,
         l1_b + (size_t)l * 1024, R_TOT, 1024, 64);
    k_gelu<<<g1((size_t)R_TOT * 1024), 256, 0, stream>>>(SCR32, SCR16,
                                                         (size_t)R_TOT * 1024);
    gemm(SCR16, 1024, l216[l], 1024, delta, 64, nullptr, 0,
         l2_b + (size_t)l * 64, R_TOT, 64, 1024);
    k_ln<<<g1(R_TOT), 256, 0, stream>>>(out32, delta,
        ln_w + (size_t)(l * 3 + 2) * 64, ln_b + (size_t)(l * 3 + 2) * 64,
        out32, out16, R_TOT);
  }

  // ---- final projection + mask -> (B,N,7) ----
  k_final<<<g1((size_t)R_TOT * POS_DIM), 256, 0, stream>>>(out32, fin_w, fin_b,
                                                           mask, outp);
}
